// MultiHeadAttention_5317169512837
// MI455X (gfx1250) — compile-verified, matrix operand order checked by reading, not run on hardware
//
#include <hip/hip_runtime.h>
#include <hip/hip_bf16.h>
#include <stdint.h>

#define NB 2
#define LQS 2048
#define LKS 2048
#define NH 16
#define PDIM 64
#define DMODEL 1024

typedef __attribute__((ext_vector_type(16))) __bf16 v16bf;
typedef __attribute__((ext_vector_type(8)))  float  v8f;

union BfOp {
  uint4  u[2];
  v16bf  v;
  __bf16 e[16];
};

union AccT {
  v8f   v;
  float f[8];
};

union Pk8 {
  uint4          u;
  __hip_bfloat16 h[8];
};

// Load 16 bf16 (elements p+0..7 and p+16..23) as two 16B chunks (global or LDS).
__device__ __forceinline__ void ld16(BfOp& o, const __hip_bfloat16* __restrict__ p) {
  const uint4* q = (const uint4*)p;
  o.u[0] = q[0];
  o.u[1] = q[2];
}

// Async copy 16 bytes global -> LDS (gfx1250, ASYNCcnt-tracked).
__device__ __forceinline__ void async_cp16(uint32_t lds_off, const void* g) {
  asm volatile("global_load_async_to_lds_b128 %0, %1, off"
               :: "v"(lds_off), "v"(g) : "memory");
}

// ---------------------------------------------------------------------------
// One-shot fp32 -> bf16 conversion (Y, X, Wq, Wk, Wv), 8 elems per thread.
// ---------------------------------------------------------------------------
__global__ __launch_bounds__(256) void cvt_bf16_kernel(
    const float* __restrict__ s, __hip_bfloat16* __restrict__ d, int n8)
{
  const int i = blockIdx.x * blockDim.x + threadIdx.x;
  if (i >= n8) return;
  const float4* p = (const float4*)s + (size_t)i * 2;
  const float4 a = p[0];
  const float4 b = p[1];
  Pk8 o;
  o.h[0] = __float2bfloat16(a.x); o.h[1] = __float2bfloat16(a.y);
  o.h[2] = __float2bfloat16(a.z); o.h[3] = __float2bfloat16(a.w);
  o.h[4] = __float2bfloat16(b.x); o.h[5] = __float2bfloat16(b.y);
  o.h[6] = __float2bfloat16(b.z); o.h[7] = __float2bfloat16(b.w);
  ((uint4*)d)[i] = o.u;
}

// ---------------------------------------------------------------------------
// Projections from pre-converted bf16: Q = Y*Wq^T, K = X*Wk^T, V = X*Wv^T
// Q,K stored bf16 as [N][H][L][PD]; V stored transposed bf16 as [N][H][PD][LK].
// One wave: 16-row x 64-col (one head) tile. grid = (Mtiles/4, H, 3)
// ---------------------------------------------------------------------------
__global__ __launch_bounds__(128) void mha_proj_kernel(
    const __hip_bfloat16* __restrict__ Yb, const __hip_bfloat16* __restrict__ Xb,
    const __hip_bfloat16* __restrict__ Wqb, const __hip_bfloat16* __restrict__ Wkb,
    const __hip_bfloat16* __restrict__ Wvb,
    __hip_bfloat16* __restrict__ qh, __hip_bfloat16* __restrict__ kh,
    __hip_bfloat16* __restrict__ vt)
{
  const int lane  = threadIdx.x & 31;
  const int wave  = threadIdx.x >> 5;
  const int mtile = blockIdx.x * 4 + wave;   // 0 .. N*L/16-1
  const int h     = blockIdx.y;              // head
  const int p     = blockIdx.z;              // 0=Q, 1=K, 2=V

  const __hip_bfloat16* src = (p == 0) ? Yb : Xb;
  const __hip_bfloat16* W   = (p == 0) ? Wqb : (p == 1) ? Wkb : Wvb;

  const int nl   = lane & 15;
  const int hi   = lane >> 4;
  const int koff = hi * 8;

  const __hip_bfloat16* arow = src + (size_t)(mtile * 16 + nl) * DMODEL;
  const __hip_bfloat16* wrow[4];
#pragma unroll
  for (int t = 0; t < 4; ++t)
    wrow[t] = W + (size_t)(h * PDIM + t * 16 + nl) * DMODEL;

  AccT c[4];
#pragma unroll
  for (int t = 0; t < 4; ++t)
#pragma unroll
    for (int i = 0; i < 8; ++i) c[t].f[i] = 0.0f;

  for (int kb = 0; kb < DMODEL; kb += 32) {
    BfOp a;
    ld16(a, arow + kb + koff);
#pragma unroll
    for (int t = 0; t < 4; ++t) {
      BfOp b;
      ld16(b, wrow[t] + kb + koff);
      c[t].v = __builtin_amdgcn_wmma_f32_16x16x32_bf16(
          false, a.v, false, b.v, (short)0, c[t].v, false, false);
    }
  }

  // C/D layout: lane holds col nl, rows r (+8 if hi)
#pragma unroll
  for (int t = 0; t < 4; ++t) {
#pragma unroll
    for (int r = 0; r < 8; ++r) {
      const int gm = mtile * 16 + r + 8 * hi;          // global A row
      const int n  = gm / LQS;
      const int l  = gm % LQS;
      const int pd = t * 16 + nl;
      const __hip_bfloat16 bv = __float2bfloat16(c[t].f[r]);
      if (p == 2) {
        vt[((size_t)(n * NH + h) * PDIM + pd) * LKS + l] = bv;
      } else {
        __hip_bfloat16* dst = (p == 0) ? qh : kh;
        dst[((size_t)(n * NH + h) * LQS + l) * PDIM + pd] = bv;
      }
    }
  }
}

// ---------------------------------------------------------------------------
// Flash attention. 4 waves/block share one (n,h); each wave owns a 16-row
// query tile. K/V chunks staged into LDS with double-buffered async copies.
// ---------------------------------------------------------------------------
__device__ __forceinline__ void stage_chunk(
    const __hip_bfloat16* __restrict__ kbase, const __hip_bfloat16* __restrict__ vbase,
    int kb, uint32_t ldsK, uint32_t ldsV, int tid)
{
  // K chunk: rows kb..kb+31 of [LK][PD] bf16 -> contiguous 4KB
  const char* gk = (const char*)(kbase + (size_t)kb * PDIM);
  async_cp16(ldsK + (uint32_t)(tid * 16),        gk + tid * 16);
  async_cp16(ldsK + (uint32_t)(tid * 16 + 2048), gk + tid * 16 + 2048);
  // V chunk: 64 rows x 32 cols from [PD][LK] bf16 (64B per row)
  int c2 = tid;
  async_cp16(ldsV + (uint32_t)((c2 >> 2) * 64 + (c2 & 3) * 16),
             vbase + (size_t)(c2 >> 2) * LKS + kb + (c2 & 3) * 8);
  c2 = tid + 128;
  async_cp16(ldsV + (uint32_t)((c2 >> 2) * 64 + (c2 & 3) * 16),
             vbase + (size_t)(c2 >> 2) * LKS + kb + (c2 & 3) * 8);
}

__global__ __launch_bounds__(128) void mha_attn_kernel(
    const __hip_bfloat16* __restrict__ qh, const __hip_bfloat16* __restrict__ kh,
    const __hip_bfloat16* __restrict__ vt, const uint8_t* __restrict__ mask,
    float* __restrict__ out)
{
  __shared__ __hip_bfloat16 bufK[2][32 * PDIM];   // 2 x 4KB
  __shared__ __hip_bfloat16 bufV[2][PDIM * 32];   // 2 x 4KB
  __shared__ __hip_bfloat16 pbuf[4][16 * 32];     // 4 x 1KB

  const int tid  = threadIdx.x;
  const int lane = tid & 31;
  const int wave = tid >> 5;
  const int qt   = blockIdx.x * 4 + wave;  // query tile 0..127
  const int h    = blockIdx.y;
  const int n    = blockIdx.z;

  const int nl   = lane & 15;
  const int hi   = lane >> 4;
  const int koff = hi * 8;

  const __hip_bfloat16* qbase = qh + ((size_t)(n * NH + h) * LQS + qt * 16) * PDIM;
  const __hip_bfloat16* kbase = kh + (size_t)(n * NH + h) * LKS * PDIM;
  const __hip_bfloat16* vbase = vt + (size_t)(n * NH + h) * PDIM * LKS;

  const uint32_t ldsK[2] = { (uint32_t)(uintptr_t)&bufK[0][0],
                             (uint32_t)(uintptr_t)&bufK[1][0] };
  const uint32_t ldsV[2] = { (uint32_t)(uintptr_t)&bufV[0][0],
                             (uint32_t)(uintptr_t)&bufV[1][0] };

  // Preload Q A-operands for both 32-wide head-dim chunks.
  BfOp qa[2];
#pragma unroll
  for (int c = 0; c < 2; ++c)
    ld16(qa[c], qbase + (size_t)nl * PDIM + c * 32 + koff);

  float rm[8], rs[8];
  AccT  o[4];
#pragma unroll
  for (int r = 0; r < 8; ++r) { rm[r] = -3.0e38f; rs[r] = 0.0f; }
#pragma unroll
  for (int t = 0; t < 4; ++t)
#pragma unroll
    for (int i = 0; i < 8; ++i) o[t].f[i] = 0.0f;

  const float sscale = 0.022097086912079608f;  // 1/sqrt(LK=2048)
  __hip_bfloat16* lp = pbuf[wave];

  // Prologue: stage chunk 0 into buffer 0.
  stage_chunk(kbase, vbase, 0, ldsK[0], ldsV[0], tid);

  const int nchunks = LKS / 32;
  for (int i = 0; i < nchunks; ++i) {
    const int kb  = i * 32;
    const int nxt = (i + 1 < nchunks) ? (kb + 32) : 0;  // wrap: dummy refill, kept for uniform waits
    stage_chunk(kbase, vbase, nxt, ldsK[(i + 1) & 1], ldsV[(i + 1) & 1], tid);
    asm volatile("s_wait_asynccnt 0x4" ::: "memory");   // chunk i resident (4 newer in flight)
    __syncthreads();                                    // all waves' async data visible

    const __hip_bfloat16* bk = bufK[i & 1];
    const __hip_bfloat16* bv = bufV[i & 1];

    // --- S = Q * K^T for two 16-col tiles (K operand from LDS) ---
    AccT s[2];
#pragma unroll
    for (int t = 0; t < 2; ++t) {
#pragma unroll
      for (int j = 0; j < 8; ++j) s[t].f[j] = 0.0f;
#pragma unroll
      for (int c = 0; c < 2; ++c) {
        BfOp kop;
        ld16(kop, bk + (t * 16 + nl) * PDIM + c * 32 + koff);
        s[t].v = __builtin_amdgcn_wmma_f32_16x16x32_bf16(
            false, qa[c].v, false, kop.v, (short)0, s[t].v, false, false);
      }
    }

    // --- online softmax (row across 16 lanes of C-layout) ---
#pragma unroll
    for (int r = 0; r < 8; ++r) {
      float a0 = s[0].f[r] * sscale;
      float a1 = s[1].f[r] * sscale;
      const int qrow = qt * 16 + r + 8 * hi;
      const uint8_t* mrow = mask + (size_t)qrow * LKS + kb + nl;
      if (mrow[0])  a0 = -1.0e10f;
      if (mrow[16]) a1 = -1.0e10f;

      float cm = fmaxf(a0, a1);
      cm = fmaxf(cm, __shfl_xor(cm, 1, 16));
      cm = fmaxf(cm, __shfl_xor(cm, 2, 16));
      cm = fmaxf(cm, __shfl_xor(cm, 4, 16));
      cm = fmaxf(cm, __shfl_xor(cm, 8, 16));

      const float nm = fmaxf(rm[r], cm);
      const float al = __expf(rm[r] - nm);
      rm[r] = nm;
      const float e0 = __expf(a0 - nm);
      const float e1 = __expf(a1 - nm);
      float sum = e0 + e1;
      sum += __shfl_xor(sum, 1, 16);
      sum += __shfl_xor(sum, 2, 16);
      sum += __shfl_xor(sum, 4, 16);
      sum += __shfl_xor(sum, 8, 16);
      rs[r] = rs[r] * al + sum;
#pragma unroll
      for (int t = 0; t < 4; ++t) o[t].f[r] *= al;

      // P tile -> LDS (row-major 16x32 bf16), C-layout writes
      const int m = r + 8 * hi;
      lp[m * 32 + nl]      = __float2bfloat16(e0);
      lp[m * 32 + 16 + nl] = __float2bfloat16(e1);
    }

    // LDS RAW on the per-wave P buffer (CDNA5 split counters)
    asm volatile("s_wait_dscnt 0x0" ::: "memory");

    // --- reload P in A-layout, then O += P * V (V operand from LDS) ---
    BfOp pa;
    ld16(pa, lp + nl * 32 + koff);
#pragma unroll
    for (int jt = 0; jt < 4; ++jt) {
      BfOp vo;
      ld16(vo, bv + (jt * 16 + nl) * 32 + koff);
      o[jt].v = __builtin_amdgcn_wmma_f32_16x16x32_bf16(
          false, pa.v, false, vo.v, (short)0, o[jt].v, false, false);
    }

    __syncthreads();  // all waves done with buf[i&1] before it is refilled
  }

  // normalize and write output [N][LQ][H*PD] fp32
#pragma unroll
  for (int jt = 0; jt < 4; ++jt) {
#pragma unroll
    for (int r = 0; r < 8; ++r) {
      const int qrow = qt * 16 + r + 8 * hi;
      const float val = o[jt].f[r] / rs[r];
      out[((size_t)n * LQS + qrow) * DMODEL + h * PDIM + jt * 16 + nl] = val;
    }
  }
}

extern "C" void kernel_launch(void* const* d_in, const int* in_sizes, int n_in,
                              void* d_out, int out_size, void* d_ws, size_t ws_size,
                              hipStream_t stream) {
  const float*   Y    = (const float*)d_in[0];
  const float*   X    = (const float*)d_in[1];
  const uint8_t* mask = (const uint8_t*)d_in[2];
  const float*   Wq   = (const float*)d_in[3];
  const float*   Wk   = (const float*)d_in[4];
  const float*   Wv   = (const float*)d_in[5];
  float*         out  = (float*)d_out;

  const size_t headElems = (size_t)NB * NH * LQS * PDIM;  // 4.19M
  const size_t seqElems  = (size_t)NB * LQS * DMODEL;     // 4.19M
  const size_t wElems    = (size_t)DMODEL * DMODEL;       // 1.05M

  __hip_bfloat16* qh  = (__hip_bfloat16*)d_ws;
  __hip_bfloat16* kh  = qh + headElems;
  __hip_bfloat16* vt  = kh + headElems;
  __hip_bfloat16* Yb  = vt + headElems;
  __hip_bfloat16* Xb  = Yb + seqElems;
  __hip_bfloat16* Wqb = Xb + seqElems;
  __hip_bfloat16* Wkb = Wqb + wElems;
  __hip_bfloat16* Wvb = Wkb + wElems;

  // One-shot bf16 conversion of all fp32 operands.
  const int seq8 = (int)(seqElems / 8), w8 = (int)(wElems / 8);
  cvt_bf16_kernel<<<(seq8 + 255) / 256, 256, 0, stream>>>(Y,  Yb,  seq8);
  cvt_bf16_kernel<<<(seq8 + 255) / 256, 256, 0, stream>>>(X,  Xb,  seq8);
  cvt_bf16_kernel<<<(w8 + 255) / 256,   256, 0, stream>>>(Wq, Wqb, w8);
  cvt_bf16_kernel<<<(w8 + 255) / 256,   256, 0, stream>>>(Wk, Wkb, w8);
  cvt_bf16_kernel<<<(w8 + 255) / 256,   256, 0, stream>>>(Wv, Wvb, w8);

  // Projections: 256 M-tiles, 4 waves/block, 16 heads, 3 projections
  dim3 gp((NB * LQS / 16) / 4, NH, 3);
  mha_proj_kernel<<<gp, 128, 0, stream>>>(Yb, Xb, Wqb, Wkb, Wvb, qh, kh, vt);

  // Attention: 128 q-tiles, 4 waves/block, per (h, n)
  dim3 ga((LQS / 16) / 4, NH, NB);
  mha_attn_kernel<<<ga, 128, 0, stream>>>(qh, kh, vt, mask, out);
}
